// GraphConstructor_35485019800147
// MI455X (gfx1250) — compile-verified, matrix operand order checked by reading, not run on hardware
//
#include <hip/hip_runtime.h>
#include <hip/hip_bf16.h>

// ---------------------------------------------------------------------------
// GraphConstructor for MI455X (gfx1250, wave32, WMMA).
//
// Roofline: dominant cost = streaming time1/time2 (503 MB fp32) once from HBM
// (~22 us @ 23.3 TB/s). All GEMMs run on v_wmma_f32_16x16x32_bf16 with
// on-the-fly f32->bf16 conversion of the streamed operand. Wt1 is transposed/
// converted to bf16 once (31.5 MB -> L2-resident at 192 MB). The big K=61440
// GEMMs are split-K (16 chunks -> 256 WGs each) accumulating with
// global_atomic_add_f32 so enough WGs are in flight to saturate HBM.
// Wave tile 32x64 (2 A-frags x 4 B-frags -> 8 WMMA per K-step of 32).
// ---------------------------------------------------------------------------

typedef __bf16 bf16_t;
typedef __attribute__((ext_vector_type(8)))  __bf16 v8bf;
typedef __attribute__((ext_vector_type(16))) __bf16 v16bf;
typedef __attribute__((ext_vector_type(8)))  float  v8f;

#define NN   1024
#define TFK  61440      // N * TF
#define HH   256
#define NF   128
#define DELTA_MIN 0.05f

// ---------------------------------------------------------------------------
// Transpose + convert f32 [R][C] -> bf16 [C][R].  32x32 tiles, 256 threads.
// ---------------------------------------------------------------------------
__global__ __launch_bounds__(256) void transpose_cvt_bf16(
    const float* __restrict__ in, bf16_t* __restrict__ out, int R, int C) {
  __shared__ float tile[32][33];
  const int tx = threadIdx.x & 31;
  const int ty = threadIdx.x >> 5;            // 0..7
  const int r0 = blockIdx.y * 32;
  const int c0 = blockIdx.x * 32;
#pragma unroll
  for (int j = 0; j < 4; ++j) {
    const int rr = ty + j * 8;
    tile[rr][tx] = in[(size_t)(r0 + rr) * C + c0 + tx];
  }
  __syncthreads();
#pragma unroll
  for (int j = 0; j < 4; ++j) {
    const int cc = ty + j * 8;
    out[(size_t)(c0 + cc) * R + r0 + tx] = (bf16_t)tile[tx][cc];
  }
}

// ---------------------------------------------------------------------------
// WMMA GEMM:  Out[M][N] (+)= epi( A[M][K(chunk)] * Bt[N][K(chunk)]^T )
//   A: f32 (cvt to bf16 on the fly) or bf16 ;  Bt: bf16, N-major (ld = K)
//   OUT_MODE: 0 = bf16 store (+bias/relu/addend), 1 = f32 store,
//             2 = split-K: global_atomic_add_f32 partial accumulate
// WG tile 128(M) x 128(N); 8 waves (4M x 2N); wave tile 32x64; K step 32.
// gridDim.z selects the K chunk (kBegin = z*Kchunk).
// Requires M%128==0, N%128==0, Kchunk%32==0.
// ---------------------------------------------------------------------------
template <bool A_F32, int OUT_MODE>
__global__ __launch_bounds__(256) void wmma_gemm(
    const void* __restrict__ Av, const bf16_t* __restrict__ Bt,
    const float* __restrict__ bias, const bf16_t* __restrict__ addend,
    void* __restrict__ Outv, int M, int N, int K, int Kchunk, int relu) {
  constexpr int LDSP = 40;    // padded bf16 stride (80 B; 16 B-aligned chunks)
  __shared__ bf16_t sA[128 * LDSP];
  __shared__ bf16_t sB[128 * LDSP];

  const int tid   = threadIdx.x;
  const int lane  = tid & 31;
  const int wave  = tid >> 5;
  const int waveM = wave & 3;        // 0..3 -> M offset *32
  const int waveN = wave >> 2;       // 0..1 -> N offset *64
  const int blockM = blockIdx.x * 128;
  const int blockN = blockIdx.y * 128;
  const int kBegin = blockIdx.z * Kchunk;
  const int kEnd   = kBegin + Kchunk;

  // global->LDS loader mapping: 2 threads per row, 16 elems each
  const int ldRow = tid >> 1;            // 0..127
  const int ldCol = (tid & 1) * 16;      // 0,16

  // fragment layout per ISA 7.12.2 (bf16 16x32 A / 32x16 B, wave32)
  const int fragRow = lane & 15;
  const int kHalf   = lane >> 4;         // 0/1
  const int k0      = kHalf * 8;         // bf16 elements

  v8f acc[2][4] = {};

  for (int kt = kBegin; kt < kEnd; kt += 32) {
    if (A_F32) {
      const float* src = (const float*)Av + (size_t)(blockM + ldRow) * K + kt + ldCol;
      float4 f0 = *(const float4*)(src);
      float4 f1 = *(const float4*)(src + 4);
      float4 f2 = *(const float4*)(src + 8);
      float4 f3 = *(const float4*)(src + 12);
      v8bf h0, h1;
      h0[0] = (bf16_t)f0.x; h0[1] = (bf16_t)f0.y; h0[2] = (bf16_t)f0.z; h0[3] = (bf16_t)f0.w;
      h0[4] = (bf16_t)f1.x; h0[5] = (bf16_t)f1.y; h0[6] = (bf16_t)f1.z; h0[7] = (bf16_t)f1.w;
      h1[0] = (bf16_t)f2.x; h1[1] = (bf16_t)f2.y; h1[2] = (bf16_t)f2.z; h1[3] = (bf16_t)f2.w;
      h1[4] = (bf16_t)f3.x; h1[5] = (bf16_t)f3.y; h1[6] = (bf16_t)f3.z; h1[7] = (bf16_t)f3.w;
      *(v8bf*)&sA[ldRow * LDSP + ldCol]     = h0;
      *(v8bf*)&sA[ldRow * LDSP + ldCol + 8] = h1;
      __builtin_prefetch(src + 32, 0, 1);    // global_prefetch_b8 (speculative)
    } else {
      const bf16_t* src = (const bf16_t*)Av + (size_t)(blockM + ldRow) * K + kt + ldCol;
      *(v8bf*)&sA[ldRow * LDSP + ldCol]     = *(const v8bf*)(src);
      *(v8bf*)&sA[ldRow * LDSP + ldCol + 8] = *(const v8bf*)(src + 8);
      __builtin_prefetch(src + 32, 0, 1);
    }
    {
      const bf16_t* src = Bt + (size_t)(blockN + ldRow) * K + kt + ldCol;
      *(v8bf*)&sB[ldRow * LDSP + ldCol]     = *(const v8bf*)(src);
      *(v8bf*)&sB[ldRow * LDSP + ldCol + 8] = *(const v8bf*)(src + 8);
      __builtin_prefetch(src + 32, 0, 1);
    }
    __syncthreads();

    v16bf afrag[2];
#pragma unroll
    for (int mi = 0; mi < 2; ++mi) {
      const bf16_t* ap = &sA[(waveM * 32 + mi * 16 + fragRow) * LDSP + k0];
      v8bf alo = *(const v8bf*)ap;
      v8bf ahi = *(const v8bf*)(ap + 16);
      afrag[mi] = __builtin_shufflevector(alo, ahi, 0, 1, 2, 3, 4, 5, 6, 7,
                                          8, 9, 10, 11, 12, 13, 14, 15);
    }
#pragma unroll
    for (int nf = 0; nf < 4; ++nf) {
      const bf16_t* bp = &sB[(waveN * 64 + nf * 16 + fragRow) * LDSP + k0];
      v8bf blo = *(const v8bf*)bp;
      v8bf bhi = *(const v8bf*)(bp + 16);
      v16bf bfrag = __builtin_shufflevector(blo, bhi, 0, 1, 2, 3, 4, 5, 6, 7,
                                            8, 9, 10, 11, 12, 13, 14, 15);
      acc[0][nf] = __builtin_amdgcn_wmma_f32_16x16x32_bf16(
          false, afrag[0], false, bfrag, (short)0, acc[0][nf], false, false);
      acc[1][nf] = __builtin_amdgcn_wmma_f32_16x16x32_bf16(
          false, afrag[1], false, bfrag, (short)0, acc[1][nf], false, false);
    }
    __syncthreads();
  }

  // Epilogue. C layout: VGPR r -> M = kHalf*8 + r, N = lane&15.
#pragma unroll
  for (int mi = 0; mi < 2; ++mi) {
    const int rowBase = blockM + waveM * 32 + mi * 16 + kHalf * 8;
#pragma unroll
    for (int nf = 0; nf < 4; ++nf) {
      const int col = blockN + waveN * 64 + nf * 16 + fragRow;
      if (OUT_MODE == 2) {
#pragma unroll
        for (int r = 0; r < 8; ++r)
          unsafeAtomicAdd(&((float*)Outv)[(size_t)(rowBase + r) * N + col],
                          acc[mi][nf][r]);
      } else {
        const float b = bias ? bias[col] : 0.0f;
#pragma unroll
        for (int r = 0; r < 8; ++r) {
          float v = acc[mi][nf][r] + b;
          if (relu) v = v > 0.0f ? v : 0.0f;
          const int row = rowBase + r;
          if (addend) v += (float)addend[(size_t)row * N + col];
          if (OUT_MODE == 1) ((float*)Outv)[(size_t)row * N + col] = v;
          else               ((bf16_t*)Outv)[(size_t)row * N + col] = (bf16_t)v;
        }
      }
    }
  }
}

// ---------------------------------------------------------------------------
// Split-K epilogue: out[m][n] = bf16(relu(acc[m][n] + bias[n])), N = 256.
// ---------------------------------------------------------------------------
__global__ __launch_bounds__(256) void bias_relu_cvt(
    const float* __restrict__ acc, const float* __restrict__ bias,
    bf16_t* __restrict__ out) {
  const int idx = blockIdx.x * 256 + threadIdx.x;
  const float v = acc[idx] + bias[idx & (HH - 1)];
  out[idx] = (bf16_t)(v > 0.0f ? v : 0.0f);
}

// ---------------------------------------------------------------------------
// Fused threshold + row softmax, in place on f32 [NN][NN]. One WG per row.
// ---------------------------------------------------------------------------
__global__ __launch_bounds__(256) void softmax_rows(float* __restrict__ x) {
  float* p = x + (size_t)blockIdx.x * NN;
  const int tid = threadIdx.x;
  const int lane = tid & 31;
  const int wv = tid >> 5;
  __shared__ float red[8];

  float v[4];
  float mx = -3.0e38f;
#pragma unroll
  for (int j = 0; j < 4; ++j) {
    float t = p[tid + j * 256];
    t = (t >= DELTA_MIN) ? t : 0.0f;
    v[j] = t;
    mx = fmaxf(mx, t);
  }
#pragma unroll
  for (int off = 16; off; off >>= 1) mx = fmaxf(mx, __shfl_xor(mx, off, 32));
  if (lane == 0) red[wv] = mx;
  __syncthreads();
  mx = red[0];
#pragma unroll
  for (int i = 1; i < 8; ++i) mx = fmaxf(mx, red[i]);

  float s = 0.0f;
#pragma unroll
  for (int j = 0; j < 4; ++j) {
    v[j] = __expf(v[j] - mx);
    s += v[j];
  }
#pragma unroll
  for (int off = 16; off; off >>= 1) s += __shfl_xor(s, off, 32);
  __syncthreads();
  if (lane == 0) red[wv] = s;
  __syncthreads();
  s = 0.0f;
#pragma unroll
  for (int i = 0; i < 8; ++i) s += red[i];
  const float inv = 1.0f / s;
#pragma unroll
  for (int j = 0; j < 4; ++j) p[tid + j * 256] = v[j] * inv;
}

// ---------------------------------------------------------------------------
// Host orchestration.
// ---------------------------------------------------------------------------
static inline size_t alignUp(size_t x) { return (x + 255) & ~(size_t)255; }

extern "C" void kernel_launch(void* const* d_in, const int* in_sizes, int n_in,
                              void* d_out, int out_size, void* d_ws, size_t ws_size,
                              hipStream_t stream) {
  const float* space = (const float*)d_in[0];   // [N][N]
  const float* time1 = (const float*)d_in[1];   // [N][N*TF]
  const float* time2 = (const float*)d_in[2];
  const float* Ws1   = (const float*)d_in[3];   // [N][H]
  const float* bs1   = (const float*)d_in[4];   // [H]
  const float* Ws2   = (const float*)d_in[5];   // [H][NF]
  const float* bs2   = (const float*)d_in[6];   // [NF]
  const float* Wt1   = (const float*)d_in[7];   // [N*TF][H]
  const float* bt1   = (const float*)d_in[8];   // [H]
  const float* Wt2   = (const float*)d_in[9];   // [H][NF]
  const float* bt2   = (const float*)d_in[10];  // [NF]
  const float* Bm    = (const float*)d_in[11];  // [NF][NF]

  // workspace carve-up
  char* ws = (char*)d_ws;
  size_t off = 0;
  bf16_t* Wt1T = (bf16_t*)(ws + off); off = alignUp(off + (size_t)HH * TFK * 2);
  bf16_t* Ws1T = (bf16_t*)(ws + off); off = alignUp(off + (size_t)HH * NN * 2);
  bf16_t* Ws2T = (bf16_t*)(ws + off); off = alignUp(off + (size_t)NF * HH * 2);
  bf16_t* Wt2T = (bf16_t*)(ws + off); off = alignUp(off + (size_t)NF * HH * 2);
  bf16_t* BT   = (bf16_t*)(ws + off); off = alignUp(off + (size_t)NF * NF * 2);
  float*  Hacc = (float*)(ws + off);  off = alignUp(off + (size_t)3 * NN * HH * 4);
  float*  Hacc_s  = Hacc;
  float*  Hacc_t1 = Hacc + (size_t)NN * HH;
  float*  Hacc_t2 = Hacc + (size_t)2 * NN * HH;
  bf16_t* H1s  = (bf16_t*)(ws + off); off = alignUp(off + (size_t)NN * HH * 2);
  bf16_t* H1t1 = (bf16_t*)(ws + off); off = alignUp(off + (size_t)NN * HH * 2);
  bf16_t* H1t2 = (bf16_t*)(ws + off); off = alignUp(off + (size_t)NN * HH * 2);
  bf16_t* sp   = (bf16_t*)(ws + off); off = alignUp(off + (size_t)NN * NF * 2);
  bf16_t* U1   = (bf16_t*)(ws + off); off = alignUp(off + (size_t)NN * NF * 2);
  bf16_t* U2   = (bf16_t*)(ws + off); off = alignUp(off + (size_t)NN * NF * 2);
  bf16_t* U1B  = (bf16_t*)(ws + off); off = alignUp(off + (size_t)NN * NF * 2);
  (void)ws_size; (void)n_in; (void)in_sizes; (void)out_size;

  // 0) zero the split-K accumulators (re-done every call: deterministic)
  hipMemsetAsync(Hacc, 0, (size_t)3 * NN * HH * 4, stream);

  // 1) pack weights: transpose + f32->bf16 (Wt1T is L2-resident, 31.5 MB)
  transpose_cvt_bf16<<<dim3(HH / 32, TFK / 32), 256, 0, stream>>>(Wt1, Wt1T, TFK, HH);
  transpose_cvt_bf16<<<dim3(HH / 32, NN / 32),  256, 0, stream>>>(Ws1, Ws1T, NN, HH);
  transpose_cvt_bf16<<<dim3(NF / 32, HH / 32),  256, 0, stream>>>(Ws2, Ws2T, HH, NF);
  transpose_cvt_bf16<<<dim3(NF / 32, HH / 32),  256, 0, stream>>>(Wt2, Wt2T, HH, NF);
  transpose_cvt_bf16<<<dim3(NF / 32, NF / 32),  256, 0, stream>>>(Bm,  BT,   NF, NF);

  // 2) layer-1 GEMMs, split-K atomic f32 accumulate (256 WGs for time GEMMs)
  wmma_gemm<true, 2><<<dim3(NN / 128, HH / 128, 1), 256, 0, stream>>>(
      space, Ws1T, nullptr, nullptr, Hacc_s, NN, HH, NN, NN, 0);
  wmma_gemm<true, 2><<<dim3(NN / 128, HH / 128, 16), 256, 0, stream>>>(
      time1, Wt1T, nullptr, nullptr, Hacc_t1, NN, HH, TFK, TFK / 16, 0);
  wmma_gemm<true, 2><<<dim3(NN / 128, HH / 128, 16), 256, 0, stream>>>(
      time2, Wt1T, nullptr, nullptr, Hacc_t2, NN, HH, TFK, TFK / 16, 0);

  // 2b) bias + ReLU + convert to bf16
  bias_relu_cvt<<<(NN * HH) / 256, 256, 0, stream>>>(Hacc_s,  bs1, H1s);
  bias_relu_cvt<<<(NN * HH) / 256, 256, 0, stream>>>(Hacc_t1, bt1, H1t1);
  bias_relu_cvt<<<(NN * HH) / 256, 256, 0, stream>>>(Hacc_t2, bt1, H1t2);

  // 3) layer-2: sp computed once (shared by U1/U2); U = relu(H1t@Wt2+bt2)+sp
  wmma_gemm<false, 0><<<dim3(NN / 128, NF / 128, 1), 256, 0, stream>>>(
      H1s, Ws2T, bs2, nullptr, sp, NN, NF, HH, HH, 1);
  wmma_gemm<false, 0><<<dim3(NN / 128, NF / 128, 1), 256, 0, stream>>>(
      H1t1, Wt2T, bt2, sp, U1, NN, NF, HH, HH, 1);
  wmma_gemm<false, 0><<<dim3(NN / 128, NF / 128, 1), 256, 0, stream>>>(
      H1t2, Wt2T, bt2, sp, U2, NN, NF, HH, HH, 1);

  // 4) bilinear: U1B = U1 @ B ; X = U1B @ U2^T (U2 row-major == Bt layout)
  wmma_gemm<false, 0><<<dim3(NN / 128, NF / 128, 1), 256, 0, stream>>>(
      U1, BT, nullptr, nullptr, U1B, NN, NF, NF, NF, 0);
  wmma_gemm<false, 1><<<dim3(NN / 128, NN / 128, 1), 256, 0, stream>>>(
      U1B, U2, nullptr, nullptr, d_out, NN, NN, NF, NF, 0);

  // 5) threshold + row softmax, in place on d_out
  softmax_rows<<<NN, 256, 0, stream>>>((float*)d_out);
}
